// SpikingMLP_52312701665562
// MI455X (gfx1250) — matrix operand order repaired and checked by reading, not compiled, not run on hardware
//
#include <hip/hip_runtime.h>
#include <stdint.h>

// ---------------- problem dims (fixed by the reference) ----------------
#define DT 4
#define DB 16
#define DN 1024
#define DC 512
#define DH 2048
#define DM (DT*DB*DN)   // 65536 rows

typedef __attribute__((ext_vector_type(16))) __bf16 v16bf;
typedef __attribute__((ext_vector_type(8)))  float  v8f;
typedef __attribute__((ext_vector_type(4)))  unsigned int uint4v;
typedef __attribute__((ext_vector_type(4)))  float float4v;
typedef __attribute__((ext_vector_type(4)))  unsigned short ushort4v;
typedef int v4i_vs __attribute__((vector_size(16)));   // matches builtin param type

union FragU { uint4v q[2]; v16bf v; };

#if defined(__has_builtin)
#  if __has_builtin(__builtin_amdgcn_global_load_async_to_lds_b128)
#    define ASYNC_LDS 1
#  endif
#  if __has_builtin(__builtin_amdgcn_s_wait_asynccnt)
#    define HAVE_WAIT_ASYNC 1
#  endif
#endif

__device__ __forceinline__ unsigned short f32_to_bf16(float f) {
  union { float f; unsigned int u; } c; c.f = f;
  unsigned int r = c.u + 0x7FFFu + ((c.u >> 16) & 1u);   // round-to-nearest-even
  return (unsigned short)(r >> 16);
}
__device__ __forceinline__ float bf16_to_f32(unsigned short h) {
  union { unsigned int u; float f; } c; c.u = ((unsigned int)h) << 16; return c.f;
}

// ---------------- f32 -> bf16 conversion (x4 vectorized) ----------------
__global__ __launch_bounds__(256)
void cvt_f32_to_bf16_x4(const float* __restrict__ in, unsigned short* __restrict__ out, int n4) {
  int i = blockIdx.x * 256 + threadIdx.x;
  if (i >= n4) return;
  float4v f = ((const float4v*)in)[i];
  ushort4v o;
  #pragma unroll
  for (int k = 0; k < 4; ++k) o[k] = f32_to_bf16(f[k]);
  ((ushort4v*)out)[i] = o;
}

// ---------------- bf16 WMMA GEMM: Out[m,n] = sum_k A[m,k]*B[n,k] ----------------
// 256 threads = 8 waves (2x4); 128x256 block tile; 64x64 wave tile (4x4 wmma frags);
// BK=32; double-buffered LDS fed by async global->LDS DMA (ASYNCcnt).
template<bool OUT_BF16>
__global__ __launch_bounds__(256)
void gemm_bf16_wmma(const unsigned short* __restrict__ A,   // [M, K] bf16
                    const unsigned short* __restrict__ Bw,  // [N, K] bf16
                    void* __restrict__ Out,                 // [M, N]
                    int K, int Ncols) {
  __shared__ unsigned short As[2][128 * 40];   // padded row stride 40 (80B)
  __shared__ unsigned short Bs[2][256 * 40];

  const int tid  = threadIdx.x;
  const int lane = tid & 31;           // wave32
  const int wave = tid >> 5;
  const int wr   = wave & 1;           // 0..1 -> 64-row halves
  const int wc   = wave >> 1;          // 0..3 -> 64-col quarters
  const size_t mBase = (size_t)blockIdx.y * 128;
  const size_t nBase = (size_t)blockIdx.x * 256;

  v8f acc[4][4];
  #pragma unroll
  for (int i = 0; i < 4; ++i)
    #pragma unroll
    for (int j = 0; j < 4; ++j)
      #pragma unroll
      for (int r = 0; r < 8; ++r) acc[i][j][r] = 0.0f;

  const int ksteps = K >> 5;

  // stage one 128x32 A tile + 256x32 B tile into LDS buffer `buf`
  auto issue_copy = [&](int buf, int kb) {
    const int k0 = kb << 5;
#if ASYNC_LDS
    #pragma unroll
    for (int i = 0; i < 2; ++i) {
      int u = tid + i * 256, row = u >> 2, cc = u & 3;
      const unsigned short* g = A + (mBase + row) * (size_t)K + k0 + cc * 8;
      unsigned short* l = &As[buf][row * 40 + cc * 8];
      __builtin_amdgcn_global_load_async_to_lds_b128(
          (__attribute__((address_space(1))) v4i_vs*)g,
          (__attribute__((address_space(3))) v4i_vs*)l, 0, 0);
    }
    #pragma unroll
    for (int i = 0; i < 4; ++i) {
      int u = tid + i * 256, row = u >> 2, cc = u & 3;
      const unsigned short* g = Bw + (nBase + row) * (size_t)K + k0 + cc * 8;
      unsigned short* l = &Bs[buf][row * 40 + cc * 8];
      __builtin_amdgcn_global_load_async_to_lds_b128(
          (__attribute__((address_space(1))) v4i_vs*)g,
          (__attribute__((address_space(3))) v4i_vs*)l, 0, 0);
    }
#else
    #pragma unroll
    for (int i = 0; i < 2; ++i) {
      int u = tid + i * 256, row = u >> 2, cc = u & 3;
      uint4v v = *(const uint4v*)(A + (mBase + row) * (size_t)K + k0 + cc * 8);
      *(uint4v*)(&As[buf][row * 40 + cc * 8]) = v;
    }
    #pragma unroll
    for (int i = 0; i < 4; ++i) {
      int u = tid + i * 256, row = u >> 2, cc = u & 3;
      uint4v v = *(const uint4v*)(Bw + (nBase + row) * (size_t)K + k0 + cc * 8);
      *(uint4v*)(&Bs[buf][row * 40 + cc * 8]) = v;
    }
#endif
  };

  auto wait_copy_barrier = [&]() {
#if ASYNC_LDS
#  if HAVE_WAIT_ASYNC
    __builtin_amdgcn_s_wait_asynccnt(0);
#  else
    asm volatile("s_wait_asynccnt 0" ::: "memory");
#  endif
#endif
    __syncthreads();
  };

  issue_copy(0, 0);
  wait_copy_barrier();

  for (int kb = 0; kb < ksteps; ++kb) {
    const int cur = kb & 1;
    if (kb + 1 < ksteps) issue_copy(cur ^ 1, kb + 1);   // DMA next tile while computing

    // A fragments (16x32 bf16 layout: 16B chunks at K = (lane/16)*8 and 16+(lane/16)*8)
    FragU af[4];
    #pragma unroll
    for (int i = 0; i < 4; ++i) {
      int r  = wr * 64 + i * 16 + (lane & 15);
      int co = (lane >> 4) * 8;
      af[i].q[0] = *(const uint4v*)(&As[cur][r * 40 + co]);
      af[i].q[1] = *(const uint4v*)(&As[cur][r * 40 + co + 16]);
    }
    // one B fragment live at a time (32x16 bf16: 16B chunks at K = (lane/16)*8 and 16+(lane/16)*8)
    #pragma unroll
    for (int j = 0; j < 4; ++j) {
      int r  = wc * 64 + j * 16 + (lane & 15);
      int co = (lane >> 4) * 8;
      FragU bfr;
      bfr.q[0] = *(const uint4v*)(&Bs[cur][r * 40 + co]);
      bfr.q[1] = *(const uint4v*)(&Bs[cur][r * 40 + co + 16]);
      #pragma unroll
      for (int i = 0; i < 4; ++i)
        acc[i][j] = __builtin_amdgcn_wmma_f32_16x16x32_bf16(
            false, af[i].v, false, bfr.v, (short)0, acc[i][j], false, false);
    }

    wait_copy_barrier();   // next buffer ready; all waves done reading current
  }

  // store: C/D 16x16 f32 layout: VGPR r -> M = r + 8*(lane/16), N = lane%16
  const int rl = (lane >> 4) * 8;
  const int cl = lane & 15;
  #pragma unroll
  for (int i = 0; i < 4; ++i)
    #pragma unroll
    for (int j = 0; j < 4; ++j) {
      size_t row0 = mBase + wr * 64 + i * 16 + rl;
      size_t col  = nBase + wc * 64 + j * 16 + cl;
      #pragma unroll
      for (int r = 0; r < 8; ++r) {
        size_t idx = (row0 + r) * (size_t)Ncols + col;
        if (OUT_BF16) ((unsigned short*)Out)[idx] = f32_to_bf16(acc[i][j][r]);
        else          ((float*)Out)[idx]          = acc[i][j][r];
      }
    }
}

// ---------------- deterministic per-channel stats: stage 1 (partials) ----------------
template<bool IN_BF16>
__global__ __launch_bounds__(256)
void col_stats_partial(const void* __restrict__ In, float* __restrict__ psum,
                       float* __restrict__ psumsq, int Mrows, int CH) {
  const int t  = threadIdx.x;
  const int ch = blockIdx.x * 64 + (t & 63);
  const int ms = t >> 6;
  const int chunk = Mrows / gridDim.y;
  const int m0 = blockIdx.y * chunk;
  float s = 0.f, q = 0.f;
  for (int m = m0 + ms; m < m0 + chunk; m += 4) {
    float v;
    if (IN_BF16) v = bf16_to_f32(((const unsigned short*)In)[(size_t)m * CH + ch]);
    else         v = ((const float*)In)[(size_t)m * CH + ch];
    s += v; q += v * v;
  }
  __shared__ float ss[256], sq[256];
  ss[t] = s; sq[t] = q;
  __syncthreads();
  if (ms == 0) {
    float S = ss[t] + ss[t + 64] + ss[t + 128] + ss[t + 192];
    float Q = sq[t] + sq[t + 64] + sq[t + 128] + sq[t + 192];
    psum[(size_t)blockIdx.y * CH + ch]   = S;
    psumsq[(size_t)blockIdx.y * CH + ch] = Q;
  }
}

// ---------------- stats stage 2: mean + rstd ----------------
__global__ __launch_bounds__(256)
void col_stats_final(const float* __restrict__ psum, const float* __restrict__ psumsq,
                     float* __restrict__ meanOut, float* __restrict__ rstdOut,
                     int CH, int SPLIT, float invM) {
  int ch = blockIdx.x * blockDim.x + threadIdx.x;
  if (ch >= CH) return;
  float s = 0.f, q = 0.f;
  for (int i = 0; i < SPLIT; ++i) {
    s += psum[(size_t)i * CH + ch];
    q += psumsq[(size_t)i * CH + ch];
  }
  float mean = s * invM;
  float var  = fmaxf(q * invM - mean * mean, 0.f);
  meanOut[ch] = mean;
  rstdOut[ch] = rsqrtf(var + 1e-5f);
}

// ---------------- BN + LIF (T=4 scan), stage 1: bf16 in-place spikes ----------------
__global__ __launch_bounds__(256)
void bn_lif_stage1(unsigned short* __restrict__ Y,
                   const float* __restrict__ mean, const float* __restrict__ rstd,
                   const float* __restrict__ g, const float* __restrict__ b,
                   int total, int CH, size_t tstride) {
  int e = blockIdx.x * 256 + threadIdx.x;
  if (e >= total) return;
  int h = e % CH;
  float mu = mean[h], rs = rstd[h], gg = g[h], bb = b[h];
  float v = 0.f;
  #pragma unroll
  for (int t = 0; t < DT; ++t) {
    size_t idx = (size_t)t * tstride + e;
    float y = (bf16_to_f32(Y[idx]) - mu) * rs * gg + bb;
    v += 0.5f * (y - v);                       // tau = 2
    float sp = (v >= 1.0f) ? 1.0f : 0.0f;      // v_th = 1
    v *= (1.0f - sp);                          // hard reset
    Y[idx] = f32_to_bf16(sp);                  // spikes {0,1}: exact in bf16
  }
}

// ---------------- BN + LIF + residual, stage 2: f32 out ----------------
__global__ __launch_bounds__(256)
void bn_lif_stage2(const float* __restrict__ Z,
                   const float* __restrict__ mean, const float* __restrict__ rstd,
                   const float* __restrict__ g, const float* __restrict__ b,
                   const float* __restrict__ x, float* __restrict__ out,
                   int total, int CH, size_t tstride) {
  int e = blockIdx.x * 256 + threadIdx.x;
  if (e >= total) return;
  int c = e % CH;
  float mu = mean[c], rs = rstd[c], gg = g[c], bb = b[c];
  float v = 0.f;
  #pragma unroll
  for (int t = 0; t < DT; ++t) {
    size_t idx = (size_t)t * tstride + e;
    float z = (Z[idx] - mu) * rs * gg + bb;
    v += 0.5f * (z - v);
    float sp = (v >= 1.0f) ? 1.0f : 0.0f;
    v *= (1.0f - sp);
    out[idx] = sp + x[idx];                    // + identity
  }
}

// ---------------- host-side launch ----------------
extern "C" void kernel_launch(void* const* d_in, const int* in_sizes, int n_in,
                              void* d_out, int out_size, void* d_ws, size_t ws_size,
                              hipStream_t stream) {
  (void)in_sizes; (void)n_in; (void)out_size; (void)ws_size;
  const float* x     = (const float*)d_in[0];   // [T,B,N,C]
  const float* fc1_w = (const float*)d_in[1];   // [H,C]
  // d_in[2] = fc1_b: cancels under train-mode BN (constant per channel)
  const float* bn1_g = (const float*)d_in[3];
  const float* bn1_b = (const float*)d_in[4];
  const float* fc2_w = (const float*)d_in[5];   // [C,H]
  // d_in[6] = fc2_b: cancels under train-mode BN
  const float* bn2_g = (const float*)d_in[7];
  const float* bn2_b = (const float*)d_in[8];
  float* out = (float*)d_out;

  // workspace carve-up (bytes)
  char* ws = (char*)d_ws;
  unsigned short* Xbf  = (unsigned short*)(ws);                    //  64 MB [M,C] bf16
  unsigned short* W1bf = (unsigned short*)(ws +  67108864);        //   2 MB [H,C] bf16
  unsigned short* W2bf = (unsigned short*)(ws +  69206016);        //   2 MB [C,H] bf16
  unsigned short* Ybf  = (unsigned short*)(ws +  71303168);        // 256 MB [M,H] bf16 (act -> spikes in place)
  float*          Zf   = (float*)(ws + 339738624);                 // 128 MB [M,C] f32
  float*          psum   = (float*)(ws + 473956352);
  float*          psumsq = psum + 64 * DH;
  float*          mean1  = psumsq + 64 * DH;
  float*          rstd1  = mean1 + DH;
  float*          mean2  = rstd1 + DH;
  float*          rstd2  = mean2 + DC;

  const int SPLIT = 64;

  // 0) bf16 conversions (vectorized x4)
  cvt_f32_to_bf16_x4<<<(DM * DC / 4 + 255) / 256, 256, 0, stream>>>(x, Xbf, DM * DC / 4);
  cvt_f32_to_bf16_x4<<<(DH * DC / 4 + 255) / 256, 256, 0, stream>>>(fc1_w, W1bf, DH * DC / 4);
  cvt_f32_to_bf16_x4<<<(DC * DH / 4 + 255) / 256, 256, 0, stream>>>(fc2_w, W2bf, DC * DH / 4);

  // 1) GEMM1: Y = X @ W1^T   [65536 x 2048], K=512
  gemm_bf16_wmma<true><<<dim3(DH / 256, DM / 128), 256, 0, stream>>>(Xbf, W1bf, Ybf, DC, DH);

  // 2) BN1 stats + BN1+LIF1 (spikes overwrite Ybf)
  col_stats_partial<true><<<dim3(DH / 64, SPLIT), 256, 0, stream>>>(Ybf, psum, psumsq, DM, DH);
  col_stats_final<<<(DH + 255) / 256, 256, 0, stream>>>(psum, psumsq, mean1, rstd1, DH, SPLIT, 1.0f / DM);
  bn_lif_stage1<<<(DB * DN * DH + 255) / 256, 256, 0, stream>>>(
      Ybf, mean1, rstd1, bn1_g, bn1_b, DB * DN * DH, DH, (size_t)DB * DN * DH);

  // 3) GEMM2: Z = spikes @ W2^T   [65536 x 512], K=2048
  gemm_bf16_wmma<false><<<dim3(DC / 256, DM / 128), 256, 0, stream>>>(Ybf, W2bf, Zf, DH, DC);

  // 4) BN2 stats + BN2+LIF2 + residual
  col_stats_partial<false><<<dim3(DC / 64, SPLIT), 256, 0, stream>>>(Zf, psum, psumsq, DM, DC);
  col_stats_final<<<(DC + 255) / 256, 256, 0, stream>>>(psum, psumsq, mean2, rstd2, DC, SPLIT, 1.0f / DM);
  bn_lif_stage2<<<(DB * DN * DC + 255) / 256, 256, 0, stream>>>(
      Zf, mean2, rstd2, bn2_g, bn2_b, x, out, DB * DN * DC, DC, (size_t)DB * DN * DC);
}
